// FrequencyFeatureExtractor_72507637891536
// MI455X (gfx1250) — matrix operand order repaired and checked by reading, not compiled, not run on hardware
//
#include <hip/hip_runtime.h>
#include <hip/hip_bf16.h>

// ---------------------------------------------------------------------------
// FrequencyFeatureExtractor for MI455X (gfx1250, wave32, WMMA + TDM)
//
//   DCT branch : mean 8x8 block -> one 8x8 DCT -> bilinear-interval matrix A
//                (21x8) -> window sums R = A * dct_img * A^T -> 3x3 tap dot.
//   FFT branch : window sums of fft2(x) expressed as two WMMA GEMMs with
//                Dirichlet kernels K[224x48] (re|im|pad):
//                  GEMM1:  QT[48x224] = (X[224x224] @ K)^T   (A-tile via TDM->LDS)
//                  GEMM2:  C2[48x48]  = K^T @ Q              (all contiguous)
//   FC         : [32x6272] @ [6272x512]^T + bias, ReLU  (WMMA)
// ---------------------------------------------------------------------------

#define IMG 224
#define BATCH 32
#define NW 21           // 7 pool windows * 3 conv taps
#define NIMG 96         // B*3 fft images
#define FEAT 6272
#define OUTC 512

typedef __attribute__((ext_vector_type(16))) _Float16 v16h;
typedef __attribute__((ext_vector_type(8)))  _Float16 v8h;
typedef __attribute__((ext_vector_type(8)))  float    v8f;
typedef __attribute__((ext_vector_type(4)))  unsigned int u32x4;
typedef __attribute__((ext_vector_type(4)))  int      i32x4;
typedef __attribute__((ext_vector_type(8)))  int      i32x8;

#if defined(__has_builtin)
#  if __has_builtin(__builtin_amdgcn_tensor_load_to_lds)
#    define HAVE_TDM 1
#  endif
#endif
#ifndef HAVE_TDM
#  define HAVE_TDM 0
#endif
#if __has_include(<hip/amd_detail/amd_gfx1250_TDM.h>)
#  define TDM_6ARG 1
#else
#  define TDM_6ARG 0
#endif

// Diagnostics (visible in compiler stderr even on success):
#if HAVE_TDM
#  if TDM_6ARG
#    warning "CDNA5 probe: tensor_load_to_lds ENABLED (6-arg therock form)"
#  else
#    warning "CDNA5 probe: tensor_load_to_lds ENABLED (5-arg ROCm form)"
#  endif
#else
#  warning "CDNA5 probe: tensor_load_to_lds NOT available - vector-copy fallback"
#endif

// ---------------- WMMA fragment helpers (16x16x32 f16, wave32) -------------
// A (16x32 row-major): lane m=lane&15, half=lane>>4
//   K = k0 + half*16 + e
__device__ inline v16h load_a_frag(const _Float16* A, int lda, int m0, int k0, int lane) {
  const _Float16* base = A + (size_t)(m0 + (lane & 15)) * lda + k0 + (lane >> 4) * 16;
  v8h lo = *(const v8h*)base;
  v8h hi = *(const v8h*)(base + 8);
  v16h a;
#pragma unroll
  for (int e = 0; e < 8; ++e) { a[e] = lo[e]; a[8 + e] = hi[e]; }
  return a;
}

__device__ inline v16h load_a_frag_lds(const _Float16* A, int k0, int lane) {
  const _Float16* base = A + (lane & 15) * IMG + k0 + (lane >> 4) * 16;
  v8h lo = *(const v8h*)base;
  v8h hi = *(const v8h*)(base + 8);
  v16h a;
#pragma unroll
  for (int e = 0; e < 8; ++e) { a[e] = lo[e]; a[8 + e] = hi[e]; }
  return a;
}

// ---------------- TDM: 2D tile (16 rows x 224 f16, stride 224) -> LDS ------
#if HAVE_TDM
__device__ inline void tdm_load_tile_2d(const void* gaddr, unsigned lds_off) {
  unsigned long long ga = (unsigned long long)(size_t)gaddr;
  u32x4 g0 = {0u, 0u, 0u, 0u};
  g0[0] = 1u;                                            // count = 1 (valid D#)
  g0[1] = lds_off;                                       // lds_addr (bytes)
  g0[2] = (unsigned)ga;                                  // global_addr[31:0]
  g0[3] = ((unsigned)(ga >> 32) & 0x1FFFFFFu) | (2u << 30);  // addr[56:32] | type=2
  i32x8 g1 = {0, 0, 0, 0, 0, 0, 0, 0};
  g1[0] = 1 << 16;                                       // data_size = 2 bytes
  g1[1] = IMG << 16;                                     // tensor_dim0 = 224
  g1[2] = 16 << 16;                                      // tensor_dim1 = 16
  g1[3] = IMG << 16;                                     // tile_dim0 = 224
  g1[4] = 16;                                            // tile_dim1 = 16 (tile_dim2=0)
  g1[5] = IMG;                                           // tensor_dim0_stride = 224
  i32x4 z4 = {0, 0, 0, 0};
  i32x8 z8 = {0, 0, 0, 0, 0, 0, 0, 0};
#if TDM_6ARG
  __builtin_amdgcn_tensor_load_to_lds(g0, g1, z4, z4, z8, 0);
#else
  __builtin_amdgcn_tensor_load_to_lds(g0, g1, z4, z4, 0);
#endif
}
#endif

// ---------------- K0: Dirichlet kernel table (f16) + bilinear sums ---------
// window w = py*3 + (dy+1): indices [32*py+dy, 32*py+32+dy) clipped to [0,224)
__global__ void k_tables(_Float16* Kf16, float* Arow) {
  int w  = blockIdx.x;                 // 0..20
  int py = w / 3;
  int dy = (w % 3) - 1;
  int lo = max(0, 32 * py + dy);
  int hi = min(IMG, 32 * py + 32 + dy);

  for (int y = threadIdx.x; y < IMG; y += blockDim.x) {
    float cr = 0.f, ci = 0.f;
    for (int u = lo; u < hi; ++u) {
      int p = (u * y) % IMG;
      float th = -6.28318530718f * (float)p / (float)IMG;
      cr += cosf(th);
      ci += sinf(th);
    }
    Kf16[w * IMG + y]        = (_Float16)cr;   // rows 0..20  : Re kernel
    Kf16[(NW + w) * IMG + y] = (_Float16)ci;   // rows 21..41 : Im kernel
  }
  if (blockIdx.x == 0) {                       // zero pad rows 42..47
    for (int i = threadIdx.x; i < 6 * IMG; i += blockDim.x)
      Kf16[42 * IMG + i] = (_Float16)0.f;
  }
  if (threadIdx.x == 0) {                      // bilinear interval sums (21x8)
    float acc[8];
#pragma unroll
    for (int i = 0; i < 8; ++i) acc[i] = 0.f;
    for (int y = lo; y < hi; ++y) {
      float src = (y + 0.5f) * (8.0f / (float)IMG) - 0.5f;
      float fl = floorf(src);
      int i0 = (int)fl;
      float t = src - fl;
      int ia = min(max(i0, 0), 7);
      int ib = min(max(i0 + 1, 0), 7);
      acc[ia] += 1.0f - t;
      acc[ib] += t;
    }
#pragma unroll
    for (int i = 0; i < 8; ++i) Arow[w * 8 + i] = acc[i];
  }
}

// ---------------- K0b: pack B1[k][n] = Kf16[n][k] into fragment order ------
// Bpack[kt][nt][lane][e], k = kt*32 + (lane>>4)*16 + e, n = nt*16 + (lane&15)
__global__ void k_pack(const _Float16* __restrict__ Kf16, _Float16* __restrict__ Bpack) {
  int i = blockIdx.x * blockDim.x + threadIdx.x;
  if (i >= 7 * 3 * 32 * 16) return;
  int e = i & 15;
  int lane = (i >> 4) & 31;
  int nt = (i >> 9) % 3;
  int kt = (i >> 9) / 3;
  int k = kt * 32 + (lane >> 4) * 16 + e;
  int n = nt * 16 + (lane & 15);
  Bpack[i] = Kf16[n * IMG + k];
}

// ---------------- conversions ----------------------------------------------
__global__ void k_cvt_x(const float* __restrict__ x, _Float16* __restrict__ xf16, int n) {
  int i = blockIdx.x * blockDim.x + threadIdx.x;
  if (i < n) xf16[i] = (_Float16)x[i];
}
__global__ void k_cvt_w(const float* __restrict__ w, _Float16* __restrict__ wf16, int n) {
  int i = blockIdx.x * blockDim.x + threadIdx.x;
  if (i < n) wf16[i] = (_Float16)w[i];
}

// ---------------- K2: mean 8x8 block -> 2D ortho DCT -----------------------
__global__ void k_dct_img(const float* __restrict__ x, float* __restrict__ dct_img) {
  int b = blockIdx.x;
  __shared__ float Msum[64];
  __shared__ float Dm[64];
  if (threadIdx.x < 64) Msum[threadIdx.x] = 0.f;
  __syncthreads();
  const float* xr = x + (size_t)b * 3 * IMG * IMG;
  for (int task = threadIdx.x; task < 64 * 27; task += blockDim.x) {
    int ij = task & 63;
    int by = task >> 6;                 // 0..26
    int i = ij >> 3, j = ij & 7;
    int y = by * 8 + i;
    float s = 0.f;
    for (int bx = 0; bx < 27; ++bx) {
      size_t off = (size_t)y * IMG + bx * 8 + j;
      s += 0.299f * xr[off] + 0.587f * xr[(size_t)IMG * IMG + off] +
           0.114f * xr[2 * (size_t)IMG * IMG + off];
    }
    atomicAdd(&Msum[ij], s);
  }
  if (threadIdx.x < 64) {               // orthonormal DCT-II matrix
    int k = threadIdx.x >> 3, m = threadIdx.x & 7;
    float d = cosf(3.14159265359f * (float)((2 * m + 1) * k) / 16.0f) * 0.5f;
    if (k == 0) d *= 0.70710678119f;
    Dm[threadIdx.x] = d;
  }
  __syncthreads();
  if (threadIdx.x < 64) {               // out = D * M * D^T (M = mean*255)
    int i = threadIdx.x >> 3, l = threadIdx.x & 7;
    float acc = 0.f;
    for (int j = 0; j < 8; ++j) {
      float t = 0.f;
      for (int k = 0; k < 8; ++k) t += Msum[j * 8 + k] * Dm[l * 8 + k];
      acc += Dm[i * 8 + j] * t;
    }
    dct_img[b * 64 + threadIdx.x] = acc * (255.0f / 729.0f);
  }
}

// ---------------- K3: DCT-branch window sums R = A * dct_img * A^T ---------
__global__ void k_rdct(const float* __restrict__ dct_img, const float* __restrict__ Arow,
                       float* __restrict__ R) {
  int b = blockIdx.x;
  __shared__ float M[64];
  if (threadIdx.x < 64) M[threadIdx.x] = dct_img[b * 64 + threadIdx.x];
  __syncthreads();
  for (int o = threadIdx.x; o < NW * NW; o += blockDim.x) {
    int wu = o / NW, wv = o % NW;
    float acc = 0.f;
    for (int i = 0; i < 8; ++i) {
      float ai = Arow[wu * 8 + i];
      for (int j = 0; j < 8; ++j) acc += ai * Arow[wv * 8 + j] * M[i * 8 + j];
    }
    R[(size_t)b * (NW * NW) + o] = acc;
  }
}

// ---------------- K4: WMMA GEMM1  QT[48x224] = (X @ K)^T  ------------------
// A-tile (16x224 f16) staged to LDS by the Tensor Data Mover.
// Dual accumulators break the WMMA->WMMA RAW chain (5-NOP hazard, ISA 7.12.1).
__global__ void k_q_gemm(const _Float16* __restrict__ Bpack,
                         const _Float16* __restrict__ Xf16,
                         _Float16* __restrict__ QT) {
  __shared__ _Float16 ldsA[16 * IMG];
  int bid = blockIdx.x;                 // NIMG * 14 * 3 blocks, 1 wave each
  int img = bid / (14 * 3);
  int rem = bid % (14 * 3);
  int mt = rem / 3, nt = rem % 3;
  int lane = threadIdx.x & 31;
  const _Float16* Xtile = Xf16 + ((size_t)img * IMG + mt * 16) * IMG;

#if HAVE_TDM
  tdm_load_tile_2d(Xtile, (unsigned)(size_t)ldsA);
  __builtin_amdgcn_s_wait_tensorcnt(0);
#else
  for (int i = lane; i < 16 * IMG / 8; i += 32)
    ((v8h*)ldsA)[i] = ((const v8h*)Xtile)[i];
#endif
  __syncthreads();

  v8f c0 = {}, c1 = {};
#pragma unroll
  for (int kt = 0; kt < 7; kt += 2) {
    v16h a = load_a_frag_lds(ldsA, kt * 32, lane);
    v16h b = *(const v16h*)(Bpack + ((kt * 3 + nt) * 32 + lane) * 16);
    c0 = __builtin_amdgcn_wmma_f32_16x16x32_f16(false, a, false, b, (short)0, c0,
                                                false, false);
    if (kt + 1 < 7) {
      v16h a2 = load_a_frag_lds(ldsA, (kt + 1) * 32, lane);
      v16h b2 = *(const v16h*)(Bpack + (((kt + 1) * 3 + nt) * 32 + lane) * 16);
      c1 = __builtin_amdgcn_wmma_f32_16x16x32_f16(false, a2, false, b2, (short)0, c1,
                                                  false, false);
    }
  }
  int n = nt * 16 + (lane & 15);
  int mb = mt * 16 + 8 * (lane >> 4);
  v8h q;
#pragma unroll
  for (int r = 0; r < 8; ++r) q[r] = (_Float16)(c0[r] + c1[r]);
  *(v8h*)(QT + ((size_t)img * 48 + n) * IMG + mb) = q;   // one b128 store
}

// ---------------- K5: WMMA GEMM2  C2[48x48] = K^T @ Q ----------------------
// B[k][n] = QT[n][k] -> contiguous in k; A = Kf16 rows -> contiguous.
__global__ void k_ws2_gemm(const _Float16* __restrict__ Kf16,
                           const _Float16* __restrict__ QT,
                           float* __restrict__ C2) {
  int bid = blockIdx.x;                 // NIMG * 3 * 3 blocks, 1 wave each
  int img = bid / 9;
  int rem = bid % 9;
  int mt = rem / 3, nt = rem % 3;
  int lane = threadIdx.x & 31;
  const _Float16* Q = QT + (size_t)img * 48 * IMG;
  int n = nt * 16 + (lane & 15);
  v8f c0 = {}, c1 = {};
#pragma unroll
  for (int kt = 0; kt < 7; kt += 2) {
    int k0 = kt * 32;
    v16h a = load_a_frag(Kf16, IMG, mt * 16, k0, lane);
    v16h b = *(const v16h*)(Q + (size_t)n * IMG + k0 + (lane >> 4) * 16);
    c0 = __builtin_amdgcn_wmma_f32_16x16x32_f16(false, a, false, b, (short)0, c0,
                                                false, false);
    if (kt + 1 < 7) {
      int k1 = (kt + 1) * 32;
      v16h a2 = load_a_frag(Kf16, IMG, mt * 16, k1, lane);
      v16h b2 = *(const v16h*)(Q + (size_t)n * IMG + k1 + (lane >> 4) * 16);
      c1 = __builtin_amdgcn_wmma_f32_16x16x32_f16(false, a2, false, b2, (short)0, c1,
                                                  false, false);
    }
  }
  float* out = C2 + (size_t)img * 48 * 48;
  int half = lane >> 4;
#pragma unroll
  for (int r = 0; r < 8; ++r)
    out[(size_t)(mt * 16 + r + 8 * half) * 48 + n] = c0[r] + c1[r];
}

// ---------------- K6: 3x3-tap contraction -> pooled features (f16) ---------
// WSre[wu][wv] = C2[wu][wv] - C2[21+wu][21+wv]
// WSim[wu][wv] = C2[wu][21+wv] + C2[21+wu][wv]
__global__ void k_pooled(const float* __restrict__ dct_w, const float* __restrict__ dct_b,
                         const float* __restrict__ fft_w, const float* __restrict__ fft_b,
                         const float* __restrict__ Rdct, const float* __restrict__ C2,
                         _Float16* __restrict__ feat) {
  int b = blockIdx.x;
  const float* R = Rdct + (size_t)b * (NW * NW);
  const float inv = 1.0f / 1024.0f;
  for (int t = threadIdx.x; t < 64 * 49; t += blockDim.x) {
    int co = t / 49, s = t % 49, py = s / 7, px = s % 7;
    // DCT branch (3 identical input channels -> sum taps over ci)
    float acc = dct_b[co];
    for (int ky = 0; ky < 3; ++ky)
      for (int kx = 0; kx < 3; ++kx) {
        float we = 0.f;
        for (int ci = 0; ci < 3; ++ci) we += dct_w[((co * 3 + ci) * 3 + ky) * 3 + kx];
        acc += we * R[(py * 3 + ky) * NW + (px * 3 + kx)] * inv;
      }
    feat[(size_t)b * FEAT + co * 49 + s] = (_Float16)acc;
    // FFT branch (re/im channel pairs)
    float acc2 = fft_b[co];
    for (int cc = 0; cc < 3; ++cc) {
      const float* C2i = C2 + (size_t)(b * 3 + cc) * 48 * 48;
      for (int ky = 0; ky < 3; ++ky)
        for (int kx = 0; kx < 3; ++kx) {
          int wu = py * 3 + ky, wv = px * 3 + kx;
          float wsre = C2i[wu * 48 + wv] - C2i[(NW + wu) * 48 + (NW + wv)];
          float wsim = C2i[wu * 48 + (NW + wv)] + C2i[(NW + wu) * 48 + wv];
          float wr  = fft_w[((co * 6 + 2 * cc) * 3 + ky) * 3 + kx];
          float wim = fft_w[((co * 6 + 2 * cc + 1) * 3 + ky) * 3 + kx];
          acc2 += (wr * wsre + wim * wsim) * inv;
        }
    }
    feat[(size_t)b * FEAT + 3136 + co * 49 + s] = (_Float16)acc2;
  }
}

// ---------------- K7: WMMA FC  out[32x512] = relu(feat @ W^T + b) ----------
__global__ void k_fc_gemm(const _Float16* __restrict__ feat, const _Float16* __restrict__ W,
                          const float* __restrict__ fc_b, float* __restrict__ out) {
  int wv = blockIdx.x * (blockDim.x >> 5) + (threadIdx.x >> 5);  // 64 waves
  int lane = threadIdx.x & 31;
  int mt = wv & 1;        // 2 M tiles (B=32)
  int nt = wv >> 1;       // 32 N tiles (512)
  int n = nt * 16 + (lane & 15);
  const _Float16* Wn = W + (size_t)n * FEAT + (lane >> 4) * 16;  // B[k][n]=W[n][k]
  v8f c0 = {}, c1 = {};
  for (int k0 = 0; k0 < FEAT; k0 += 64) {      // FEAT = 98 * 64
    v16h a0 = load_a_frag(feat, FEAT, mt * 16, k0, lane);
    __builtin_prefetch(Wn + k0 + 512, 0, 1);   // global_prefetch_b8
    v16h b0 = *(const v16h*)(Wn + k0);
    c0 = __builtin_amdgcn_wmma_f32_16x16x32_f16(false, a0, false, b0, (short)0, c0,
                                                false, false);
    v16h a1 = load_a_frag(feat, FEAT, mt * 16, k0 + 32, lane);
    v16h b1 = *(const v16h*)(Wn + k0 + 32);
    c1 = __builtin_amdgcn_wmma_f32_16x16x32_f16(false, a1, false, b1, (short)0, c1,
                                                false, false);
  }
  float bias = fc_b[n];
  int half = lane >> 4;
#pragma unroll
  for (int r = 0; r < 8; ++r) {
    int m = mt * 16 + r + 8 * half;
    float v = c0[r] + c1[r] + bias;
    out[(size_t)m * OUTC + n] = v > 0.f ? v : 0.f;
  }
}

// ---------------------------------------------------------------------------
extern "C" void kernel_launch(void* const* d_in, const int* in_sizes, int n_in,
                              void* d_out, int out_size, void* d_ws, size_t ws_size,
                              hipStream_t stream) {
  (void)in_sizes; (void)n_in; (void)out_size; (void)ws_size;
  const float* x     = (const float*)d_in[0];
  const float* dct_w = (const float*)d_in[1];
  const float* dct_b = (const float*)d_in[2];
  const float* fft_w = (const float*)d_in[3];
  const float* fft_b = (const float*)d_in[4];
  const float* fc_w  = (const float*)d_in[5];
  const float* fc_b  = (const float*)d_in[6];
  float* out = (float*)d_out;

  char* ws = (char*)d_ws;
  size_t off = 0;
  auto take = [&](size_t bytes) -> void* {
    void* p = ws + off;
    off = (off + bytes + 255) & ~(size_t)255;
    return p;
  };
  float*     dct_img = (float*)take((size_t)BATCH * 64 * 4);
  _Float16*  Kf16    = (_Float16*)take((size_t)48 * IMG * 2);
  float*     Arow    = (float*)take((size_t)NW * 8 * 4);
  _Float16*  Bpack   = (_Float16*)take((size_t)7 * 3 * 32 * 16 * 2);
  float*     Rdct    = (float*)take((size_t)BATCH * NW * NW * 4);
  _Float16*  Xf16    = (_Float16*)take((size_t)NIMG * IMG * IMG * 2);
  _Float16*  QT      = (_Float16*)take((size_t)NIMG * 48 * IMG * 2);
  float*     C2      = (float*)take((size_t)NIMG * 48 * 48 * 4);
  _Float16*  feat    = (_Float16*)take((size_t)BATCH * FEAT * 2);
  _Float16*  Wf16    = (_Float16*)take((size_t)OUTC * FEAT * 2);

  const int nx = NIMG * IMG * IMG;       // 4,816,896
  const int nw = OUTC * FEAT;            // 3,211,264
  const int npack = 7 * 3 * 32 * 16;     // 10,752

  k_tables<<<NW, 256, 0, stream>>>(Kf16, Arow);
  k_pack<<<(npack + 255) / 256, 256, 0, stream>>>(Kf16, Bpack);
  k_cvt_x<<<(nx + 255) / 256, 256, 0, stream>>>(x, Xf16, nx);
  k_cvt_w<<<(nw + 255) / 256, 256, 0, stream>>>(fc_w, Wf16, nw);
  k_dct_img<<<BATCH, 256, 0, stream>>>(x, dct_img);
  k_rdct<<<BATCH, 256, 0, stream>>>(dct_img, Arow, Rdct);
  k_q_gemm<<<NIMG * 14 * 3, 32, 0, stream>>>(Bpack, Xf16, QT);
  k_ws2_gemm<<<NIMG * 9, 32, 0, stream>>>(Kf16, QT, C2);
  k_pooled<<<BATCH, 256, 0, stream>>>(dct_w, dct_b, fft_w, fft_b, Rdct, C2, feat);
  k_fc_gemm<<<8, 256, 0, stream>>>(feat, Wf16, fc_b, out);
}
